// HybridVoxelNeuralField_8744553414866
// MI455X (gfx1250) — matrix operand order repaired and checked by reading, not compiled, not run on hardware
//
#include <hip/hip_runtime.h>

// CDNA5 / gfx1250 fused voxel-grid trilinear sample + 4-layer MLP.
// wave32; WMMA f16 (f32 accumulate) for the MLP; per-wave LDS staging.

typedef _Float16 v8h  __attribute__((ext_vector_type(8)));
typedef _Float16 v16h __attribute__((ext_vector_type(16)));
typedef float    v8f  __attribute__((ext_vector_type(8)));

#define GRID_R    128
#define DHW_SHIFT 21        // 128^3 = 2^21 elements per channel slice
#define ROW_BYTES 80        // 64B of f16 features + 16B pad (bank spread)
#define WAVE_LDS  (32 * ROW_BYTES)

__global__ __launch_bounds__(256)
void voxfield_fused(const float* __restrict__ coord,
                    const float* __restrict__ grid,
                    const float* __restrict__ W1, const float* __restrict__ b1,
                    const float* __restrict__ W2, const float* __restrict__ b2,
                    const float* __restrict__ W3, const float* __restrict__ b3,
                    const float* __restrict__ W4, const float* __restrict__ b4,
                    float* __restrict__ out)
{
    __shared__ alignas(16) unsigned char lds[8 * WAVE_LDS];

    const int lane    = threadIdx.x & 31;
    const int wave    = threadIdx.x >> 5;
    const int col     = lane & 15;
    const int hiHalf  = lane >> 4;
    const unsigned waveBase = (unsigned)wave * WAVE_LDS;

    // ---- preload weights in WMMA B-operand layout (B = W^T, K x N, f16) ----
    // 16-bit B 32x16 -> lane L holds column (L&15); in this code's (non-ISA) order element e carries
    // K = (L>>4)*16 + e, i.e. a contiguous 16-float half-row of W.
    auto loadB = [&](const float* __restrict__ W, int row) -> v16h {
        v16h b;
        const float* wr = W + row * 32 + hiHalf * 16;
        #pragma unroll
        for (int e = 0; e < 16; ++e) b[e] = (_Float16)wr[e];
        return b;
    };
    v16h B1a = loadB(W1, col), B1b = loadB(W1, col + 16);
    v16h B2a = loadB(W2, col), B2b = loadB(W2, col + 16);
    v16h B3a = loadB(W3, col), B3b = loadB(W3, col + 16);
    v16h B4;
    #pragma unroll
    for (int e = 0; e < 16; ++e) B4[e] = (_Float16)0.0f;
    if (col < 4) B4 = loadB(W4, col);          // OUT=4, pad columns 4..15 with 0

    const float bias1a = b1[col], bias1b = b1[col + 16];
    const float bias2a = b2[col], bias2b = b2[col + 16];
    const float bias3a = b3[col], bias3b = b3[col + 16];
    const float bias4  = (col < 4) ? b4[col] : 0.0f;

    // ---- phase 1: trilinear gather, one point per lane ----
    const long long pt = (long long)blockIdx.x * 256 + threadIdx.x;
    const float cx = coord[pt * 3 + 0];
    const float cy = coord[pt * 3 + 1];
    const float cz = coord[pt * 3 + 2];

    auto axis = [&](float c, int* idx, float* w) {
        float ic = (c + 1.0f) * 64.0f - 0.5f;   // ((c+1)*128 - 1) * 0.5
        float f0 = floorf(ic);
        int   i0 = (int)f0;
        float fr = ic - f0;
        int   i1 = i0 + 1;
        w[0] = (1.0f - fr) * ((i0 >= 0 && i0 < GRID_R) ? 1.0f : 0.0f);
        w[1] = fr          * ((i1 >= 0 && i1 < GRID_R) ? 1.0f : 0.0f);
        idx[0] = i0 < 0 ? 0 : (i0 > GRID_R - 1 ? GRID_R - 1 : i0);
        idx[1] = i1 < 0 ? 0 : (i1 > GRID_R - 1 ? GRID_R - 1 : i1);
    };
    int xs[2], ys[2], zs[2];
    float wx[2], wy[2], wz[2];
    axis(cx, xs, wx);
    axis(cy, ys, wy);
    axis(cz, zs, wz);

    float acc[32];
    #pragma unroll
    for (int c = 0; c < 32; ++c) acc[c] = 0.0f;

    #pragma unroll
    for (int dz = 0; dz < 2; ++dz)
      #pragma unroll
      for (int dy = 0; dy < 2; ++dy)
        #pragma unroll
        for (int dx = 0; dx < 2; ++dx) {
            const float w = wz[dz] * wy[dy] * wx[dx];
            const float* gp =
                grid + (size_t)((zs[dz] * GRID_R + ys[dy]) * GRID_R + xs[dx]);
            #pragma unroll
            for (int c = 0; c < 32; ++c)
                acc[c] = fmaf(gp[(size_t)c << DHW_SHIFT], w, acc[c]);
        }

    // f32 -> f16 features into this wave's LDS tile (row = wave-local point)
    {
        unsigned char* rp = lds + waveBase + (unsigned)lane * ROW_BYTES;
        #pragma unroll
        for (int ch = 0; ch < 4; ++ch) {
            v8h v;
            #pragma unroll
            for (int e = 0; e < 8; ++e) v[e] = (_Float16)acc[ch * 8 + e];
            *(v8h*)(rp + ch * 16) = v;
        }
    }
    // wave-private LDS region: same-wave DS ordering is enough
    asm volatile("s_wait_dscnt 0x0" ::: "memory");

    // ---- phase 2: MLP on WMMA pipes, two 16-row M-tiles per wave ----
    // A as loaded here (16-bit 16x32, same k order as B, not ISA order): lane group 0 holds K 0..15,
    // group 1 holds K 16..31 -> two ds_load_b128 per lane.
    auto loadA = [&](int t) -> v16h {
        const unsigned char* rp = lds + waveBase +
            (unsigned)(t * 16 + col) * ROW_BYTES + (unsigned)hiHalf * 32;
        v8h lo = *(const v8h*)(rp);
        v8h hi = *(const v8h*)(rp + 16);
        v16h a;
        #pragma unroll
        for (int e = 0; e < 8; ++e) { a[e] = lo[e]; a[e + 8] = hi[e]; }
        return a;
    };

    // D layout (32-bit 16x16): VGPR v -> row v + 8*(lane>>4), col = lane&15.
    auto storeH = [&](int t, v8f d0, v8f d1, float ba, float bb) {
        unsigned char* base = lds + waveBase;
        #pragma unroll
        for (int v = 0; v < 8; ++v) {
            const unsigned row = (unsigned)(t * 16 + v + 8 * hiHalf);
            float h0 = fmaxf(d0[v] + ba, 0.0f);
            float h1 = fmaxf(d1[v] + bb, 0.0f);
            *(_Float16*)(base + row * ROW_BYTES + (unsigned)col * 2)        = (_Float16)h0;
            *(_Float16*)(base + row * ROW_BYTES + (unsigned)(col + 16) * 2) = (_Float16)h1;
        }
        asm volatile("s_wait_dscnt 0x0" ::: "memory");
    };

    const v8f zeroC = {};

    #pragma unroll
    for (int t = 0; t < 2; ++t) {
        v16h A = loadA(t);
        v8f d0 = __builtin_amdgcn_wmma_f32_16x16x32_f16(false, A, false, B1a,
                                                        (short)0, zeroC, false, false);
        v8f d1 = __builtin_amdgcn_wmma_f32_16x16x32_f16(false, A, false, B1b,
                                                        (short)0, zeroC, false, false);
        storeH(t, d0, d1, bias1a, bias1b);

        A = loadA(t);
        d0 = __builtin_amdgcn_wmma_f32_16x16x32_f16(false, A, false, B2a,
                                                    (short)0, zeroC, false, false);
        d1 = __builtin_amdgcn_wmma_f32_16x16x32_f16(false, A, false, B2b,
                                                    (short)0, zeroC, false, false);
        storeH(t, d0, d1, bias2a, bias2b);

        A = loadA(t);
        d0 = __builtin_amdgcn_wmma_f32_16x16x32_f16(false, A, false, B3a,
                                                    (short)0, zeroC, false, false);
        d1 = __builtin_amdgcn_wmma_f32_16x16x32_f16(false, A, false, B3b,
                                                    (short)0, zeroC, false, false);
        storeH(t, d0, d1, bias3a, bias3b);

        A = loadA(t);
        v8f d4 = __builtin_amdgcn_wmma_f32_16x16x32_f16(false, A, false, B4,
                                                        (short)0, zeroC, false, false);
        if (col < 4) {
            #pragma unroll
            for (int v = 0; v < 8; ++v) {
                const long long gp = (long long)blockIdx.x * 256 + wave * 32 +
                                     t * 16 + v + 8 * hiHalf;
                out[gp * 4 + col] = d4[v] + bias4;
            }
        }
    }
}

extern "C" void kernel_launch(void* const* d_in, const int* in_sizes, int n_in,
                              void* d_out, int out_size, void* d_ws, size_t ws_size,
                              hipStream_t stream) {
    const float* coord = (const float*)d_in[0];
    const float* gridp = (const float*)d_in[1];
    const float* W1 = (const float*)d_in[2]; const float* b1 = (const float*)d_in[3];
    const float* W2 = (const float*)d_in[4]; const float* b2 = (const float*)d_in[5];
    const float* W3 = (const float*)d_in[6]; const float* b3 = (const float*)d_in[7];
    const float* W4 = (const float*)d_in[8]; const float* b4 = (const float*)d_in[9];
    float* out = (float*)d_out;

    const int N = in_sizes[0] / 3;          // 1048576, divisible by 256
    const int blocks = N / 256;             // 256 points per block (8 waves x 32)
    voxfield_fused<<<blocks, 256, 0, stream>>>(coord, gridp,
                                               W1, b1, W2, b2, W3, b3, W4, b4,
                                               out);
}
